// SelfMultiheadAttn_8881992368301
// MI455X (gfx1250) — matrix operand order repaired and checked by reading, not compiled, not run on hardware
//
#include <hip/hip_runtime.h>

// ---------------------------------------------------------------------------
// Self multi-head attention, bf16 WMMA pipeline for gfx1250 (MI455X).
// T=2048, B=4, C=1024, H=16, D=64.
// ---------------------------------------------------------------------------

#define SEQ_T   2048
#define BATCH_B 4
#define EMBED_C 1024
#define HEADS_H 16
#define HDIM_D  64
#define MROWS   (SEQ_T * BATCH_B)   // 8192

typedef __attribute__((ext_vector_type(16))) __bf16 v16bf;
typedef __attribute__((ext_vector_type(8)))  float  v8f;

// ---- WMMA helpers ----------------------------------------------------------

__device__ __forceinline__ v8f wmma_bf16(v16bf a, v16bf b, v8f c) {
  return __builtin_amdgcn_wmma_f32_16x16x32_bf16(
      /*neg_a=*/false, a, /*neg_b=*/false, b,
      /*c_mod=*/(short)0, c, /*reuse_a=*/false, /*reuse_b=*/false);
}

// A fragment (16x32, 16-bit): lane = M row (lane&15); elements 0..15 hold
// K = k0 + half*16 .. +15.
__device__ __forceinline__ v16bf load_a_frag(const __bf16* __restrict__ arow,
                                             int k0, int half) {
  v16bf a;
#pragma unroll
  for (int i = 0; i < 8; ++i) {
    a[i]     = arow[k0 + half * 16 + i];
    a[i + 8] = arow[k0 + half * 16 + 8 + i];
  }
  return a;
}

// B fragment (32x16, 16-bit): lane = N col (lane&15); elements 0..15 hold
// K = k0 + half*16 .. +15.  `brow` points at element (n_row, 0) of a row-major
// [N x K] matrix, so the 16 loads are contiguous (32 bytes -> 2x b128).
__device__ __forceinline__ v16bf load_b_frag(const __bf16* __restrict__ brow,
                                             int k0, int half) {
  v16bf b;
#pragma unroll
  for (int i = 0; i < 16; ++i) b[i] = brow[k0 + half * 16 + i];
  return b;
}

// ---- Stage 0: fp32 -> bf16 conversion --------------------------------------

__global__ void cvt_f32_bf16(const float* __restrict__ src,
                             __bf16* __restrict__ dst) {
  size_t i = ((size_t)blockIdx.x * blockDim.x + threadIdx.x) * 4;
  float4 f = *(const float4*)(src + i);
  dst[i + 0] = (__bf16)f.x;
  dst[i + 1] = (__bf16)f.y;
  dst[i + 2] = (__bf16)f.z;
  dst[i + 3] = (__bf16)f.w;
}

// ---- Stage 1: fused QKV projection  qkv = X @ Wqkv^T + bias ----------------
// Wave register tile: 32 (M) x 64 (N): 2 A-frags x 4 B-frags -> 8 WMMA / k-step.
// A wave's 64-wide N strip is 64-aligned, so it lies inside exactly one of
// {q,k,v} and exactly one head: seg/h/dst/scale are wave-uniform scalars.

__global__ __launch_bounds__(256)
void qkv_gemm(const __bf16* __restrict__ X, const __bf16* __restrict__ W,
              const float* __restrict__ bias,
              __bf16* __restrict__ qd, __bf16* __restrict__ kd,
              __bf16* __restrict__ vd) {
  const int K = EMBED_C;
  const int NTG = (3 * EMBED_C) / 64;                  // 48 N-strips
  int wave = ((int)blockIdx.x * (int)blockDim.x + (int)threadIdx.x) >> 5;
  int lane = threadIdx.x & 31;
  int half = lane >> 4, lm = lane & 15;
  int mt  = wave / NTG;                                // 0..255 (32-row tiles)
  int ntg = wave % NTG;
  int m0 = mt * 32, n0 = ntg * 64;

  const __bf16* ar0 = X + (size_t)(m0 + lm) * K;
  const __bf16* ar1 = X + (size_t)(m0 + 16 + lm) * K;
  const __bf16* br[4];
#pragma unroll
  for (int t = 0; t < 4; ++t) br[t] = W + (size_t)(n0 + t * 16 + lm) * K;

  v8f acc[2][4] = {};
  for (int k0 = 0; k0 < K; k0 += 32) {
    __builtin_prefetch(ar0 + k0 + 128, 0, 3);
    __builtin_prefetch(ar1 + k0 + 128, 0, 3);
    v16bf a0 = load_a_frag(ar0, k0, half);
    v16bf a1 = load_a_frag(ar1, k0, half);
#pragma unroll
    for (int t = 0; t < 4; ++t) {
      __builtin_prefetch(br[t] + k0 + 128, 0, 3);
      v16bf b = load_b_frag(br[t], k0, half);
      acc[0][t] = wmma_bf16(a0, b, acc[0][t]);
      acc[1][t] = wmma_bf16(a1, b, acc[1][t]);
    }
  }

  // Wave-uniform destination selection (no per-lane branching).
  const int seg  = n0 >> 10;                 // 0=q 1=k 2=v
  const int h    = (n0 & 1023) >> 6;         // head index (strip == one head)
  __bf16* dst    = (seg == 0) ? qd : (seg == 1) ? kd : vd;
  const float sc = (seg == 0) ? 0.125f : 1.0f;   // q pre-scale 1/sqrt(64)

#pragma unroll
  for (int t = 0; t < 4; ++t) {
    float bv = bias[n0 + t * 16 + lm];
    int d = t * 16 + lm;
#pragma unroll
    for (int mi = 0; mi < 2; ++mi) {
#pragma unroll
      for (int r = 0; r < 8; ++r) {
        int m  = m0 + mi * 16 + half * 8 + r;
        int tt = m >> 2;                     // row m = t*B + b, B=4
        int bb = m & 3;
        size_t idx = (((size_t)(bb * HEADS_H + h)) * SEQ_T + tt) * HDIM_D + d;
        dst[idx] = (__bf16)((acc[mi][t][r] + bv) * sc);
      }
    }
  }
}

// ---- Stage 2: flash attention ----------------------------------------------
// One wave per (b,h, 16-row q block).  Online softmax over S in chunks of 32.
// q/k/v bf16 [B*H][T][D]; ctx bf16 [T][B][C].

__global__ __launch_bounds__(256)
void attn(const __bf16* __restrict__ q, const __bf16* __restrict__ k,
          const __bf16* __restrict__ v, __bf16* __restrict__ ctx) {
  __shared__ __bf16 pbuf[8][16][32];          // per-wave P-transpose staging

  int wib  = threadIdx.x >> 5;
  int lane = threadIdx.x & 31;
  int half = lane >> 4, lm = lane & 15;
  int wave = (int)blockIdx.x * 8 + wib;
  const int QB = SEQ_T / 16;                  // 128 q blocks
  int bh = wave / QB;
  int qb = wave % QB;

  const __bf16* qp = q + (size_t)bh * SEQ_T * HDIM_D;
  const __bf16* kp = k + (size_t)bh * SEQ_T * HDIM_D;
  const __bf16* vp = v + (size_t)bh * SEQ_T * HDIM_D;

  // Q tile 16x64 -> two resident A fragments (K=0..31, 32..63)
  const __bf16* qrow = qp + (size_t)(qb * 16 + lm) * HDIM_D;
  v16bf aq0 = load_a_frag(qrow, 0, half);
  v16bf aq1 = load_a_frag(qrow, 32, half);

  v8f o0 = {}, o1 = {}, o2 = {}, o3 = {};
  float mst[8], lst[8];
#pragma unroll
  for (int r = 0; r < 8; ++r) { mst[r] = -INFINITY; lst[r] = 0.f; }

  for (int s0 = 0; s0 < SEQ_T; s0 += 32) {
    // scores 16x32 = Q (16x64) @ K^T (64x32): 4 WMMAs
    v8f sc0 = {}, sc1 = {};
    {
      const __bf16* kr0 = kp + (size_t)(s0 + lm) * HDIM_D;
      const __bf16* kr1 = kp + (size_t)(s0 + 16 + lm) * HDIM_D;
      __builtin_prefetch(kr0 + 32 * HDIM_D, 0, 3);
      sc0 = wmma_bf16(aq0, load_b_frag(kr0, 0, half), sc0);
      sc0 = wmma_bf16(aq1, load_b_frag(kr0, 32, half), sc0);
      sc1 = wmma_bf16(aq0, load_b_frag(kr1, 0, half), sc1);
      sc1 = wmma_bf16(aq1, load_b_frag(kr1, 32, half), sc1);
    }

    // online softmax; D layout: VGPR r / half -> row half*8+r, lane -> col
#pragma unroll
    for (int r = 0; r < 8; ++r) {
      float vmax = fmaxf(sc0[r], sc1[r]);
#pragma unroll
      for (int off = 8; off >= 1; off >>= 1)
        vmax = fmaxf(vmax, __shfl_xor(vmax, off, 32));
      float mnew = fmaxf(mst[r], vmax);
      float corr = __expf(mst[r] - mnew);
      float p0 = __expf(sc0[r] - mnew);
      float p1 = __expf(sc1[r] - mnew);
      float psum = p0 + p1;
#pragma unroll
      for (int off = 8; off >= 1; off >>= 1)
        psum += __shfl_xor(psum, off, 32);
      lst[r] = lst[r] * corr + psum;
      mst[r] = mnew;
      o0[r] *= corr; o1[r] *= corr; o2[r] *= corr; o3[r] *= corr;
      pbuf[wib][half * 8 + r][lm]      = (__bf16)p0;
      pbuf[wib][half * 8 + r][16 + lm] = (__bf16)p1;
    }
    __syncthreads();

    // P (16x32) back out of LDS in A-fragment layout
    v16bf ap;
#pragma unroll
    for (int i = 0; i < 8; ++i) {
      ap[i]     = pbuf[wib][lm][half * 16 + i];
      ap[i + 8] = pbuf[wib][lm][half * 16 + 8 + i];
    }
    __syncthreads();

    // O += P @ V : V chunk 32x64, B-fragment elements walk S (stride D)
    const __bf16* vbase = vp + (size_t)(s0 + half * 16) * HDIM_D;
    v16bf bv;
#pragma unroll
    for (int j = 0; j < 4; ++j) {
#pragma unroll
      for (int i = 0; i < 16; ++i)
        bv[i] = vbase[(size_t)i * HDIM_D + j * 16 + lm];
      if (j == 0)      o0 = wmma_bf16(ap, bv, o0);
      else if (j == 1) o1 = wmma_bf16(ap, bv, o1);
      else if (j == 2) o2 = wmma_bf16(ap, bv, o2);
      else             o3 = wmma_bf16(ap, bv, o3);
    }
  }

  // normalize and write ctx [T][B][C], c = h*64 + d
  int bb = bh >> 4, h = bh & 15;
#pragma unroll
  for (int r = 0; r < 8; ++r) {
    int t = qb * 16 + half * 8 + r;
    float inv = 1.f / lst[r];
    size_t base = ((size_t)t * BATCH_B + bb) * EMBED_C + h * HDIM_D;
    ctx[base +  0 + lm] = (__bf16)(o0[r] * inv);
    ctx[base + 16 + lm] = (__bf16)(o1[r] * inv);
    ctx[base + 32 + lm] = (__bf16)(o2[r] * inv);
    ctx[base + 48 + lm] = (__bf16)(o3[r] * inv);
  }
}

// ---- Stage 3: out projection  out = ctx @ Wout^T + bias (fp32 out) ---------
// Same 32x64 register tile as stage 1; plain fp32 store to d_out.

__global__ __launch_bounds__(256)
void oproj_gemm(const __bf16* __restrict__ A, const __bf16* __restrict__ W,
                const float* __restrict__ bias, float* __restrict__ out) {
  const int K = EMBED_C;
  const int NTG = EMBED_C / 64;                        // 16 N-strips
  int wave = ((int)blockIdx.x * (int)blockDim.x + (int)threadIdx.x) >> 5;
  int lane = threadIdx.x & 31;
  int half = lane >> 4, lm = lane & 15;
  int mt  = wave / NTG;                                // 0..255
  int ntg = wave % NTG;
  int m0 = mt * 32, n0 = ntg * 64;

  const __bf16* ar0 = A + (size_t)(m0 + lm) * K;
  const __bf16* ar1 = A + (size_t)(m0 + 16 + lm) * K;
  const __bf16* br[4];
#pragma unroll
  for (int t = 0; t < 4; ++t) br[t] = W + (size_t)(n0 + t * 16 + lm) * K;

  v8f acc[2][4] = {};
  for (int k0 = 0; k0 < K; k0 += 32) {
    __builtin_prefetch(ar0 + k0 + 128, 0, 3);
    __builtin_prefetch(ar1 + k0 + 128, 0, 3);
    v16bf a0 = load_a_frag(ar0, k0, half);
    v16bf a1 = load_a_frag(ar1, k0, half);
#pragma unroll
    for (int t = 0; t < 4; ++t) {
      __builtin_prefetch(br[t] + k0 + 128, 0, 3);
      v16bf b = load_b_frag(br[t], k0, half);
      acc[0][t] = wmma_bf16(a0, b, acc[0][t]);
      acc[1][t] = wmma_bf16(a1, b, acc[1][t]);
    }
  }

#pragma unroll
  for (int t = 0; t < 4; ++t) {
    int n = n0 + t * 16 + lm;
    float bv = bias[n];
#pragma unroll
    for (int mi = 0; mi < 2; ++mi) {
#pragma unroll
      for (int r = 0; r < 8; ++r) {
        int m = m0 + mi * 16 + half * 8 + r;
        out[(size_t)m * EMBED_C + n] = acc[mi][t][r] + bv;
      }
    }
  }
}

// ---------------------------------------------------------------------------

extern "C" void kernel_launch(void* const* d_in, const int* in_sizes, int n_in,
                              void* d_out, int out_size, void* d_ws,
                              size_t ws_size, hipStream_t stream) {
  const float* query = (const float*)d_in[0];  // [T,B,C]
  const float* wqkv  = (const float*)d_in[1];  // [3C,C]
  const float* bqkv  = (const float*)d_in[2];  // [3C]
  const float* wout  = (const float*)d_in[3];  // [C,C]
  const float* bout  = (const float*)d_in[4];  // [C]
  float* out = (float*)d_out;

  char* ws = (char*)d_ws;
  const size_t MB = 1u << 20;
  __bf16* Xbf    = (__bf16*)(ws + 0 * MB);    // 16 MiB: X [8192,1024]
  __bf16* Wqkvbf = (__bf16*)(ws + 16 * MB);   //  6 MiB
  __bf16* Woutbf = (__bf16*)(ws + 22 * MB);   //  2 MiB
  __bf16* qbf    = (__bf16*)(ws + 24 * MB);   // 16 MiB [B,H,T,D]
  __bf16* kbf    = (__bf16*)(ws + 40 * MB);   // 16 MiB
  __bf16* vbf    = (__bf16*)(ws + 56 * MB);   // 16 MiB
  __bf16* ctxbf  = (__bf16*)(ws + 72 * MB);   // 16 MiB [T,B,C]

  // Stage 0: conversions (4 elems / thread)
  cvt_f32_bf16<<<(MROWS * EMBED_C) / 1024, 256, 0, stream>>>(query, Xbf);
  cvt_f32_bf16<<<(3 * EMBED_C * EMBED_C) / 1024, 256, 0, stream>>>(wqkv, Wqkvbf);
  cvt_f32_bf16<<<(EMBED_C * EMBED_C) / 1024, 256, 0, stream>>>(wout, Woutbf);

  // Stage 1: QKV projection. waves = (8192/32) * 48 = 12288 -> 1536 blocks
  qkv_gemm<<<1536, 256, 0, stream>>>(Xbf, Wqkvbf, bqkv, qbf, kbf, vbf);

  // Stage 2: attention. waves = 64 * 128 = 8192 -> 1024 blocks
  attn<<<1024, 256, 0, stream>>>(qbf, kbf, vbf, ctxbf);

  // Stage 3: out projection. waves = (8192/32) * 16 = 4096 -> 512 blocks
  oproj_gemm<<<512, 256, 0, stream>>>(ctxbf, Woutbf, bout, out);
}
